// VQEncoder_53901839565724
// MI455X (gfx1250) — compile-verified, matrix operand order checked by reading, not run on hardware
//
#include <hip/hip_runtime.h>
#include <hip/hip_bf16.h>
#include <math.h>

typedef __attribute__((ext_vector_type(16))) _Float16 v16h;
typedef __attribute__((ext_vector_type(8)))  float    v8f;

#define D_MODEL 512
#define SEQ     512
#define NTOK    4096      // B*S
#define NHEAD_T 64        // B*H flattened heads
#define DPH     64
#define FFN_DIM 2048
#define KCODES  16384
#define N_LAYERS 4

// K-index pattern for 16-bit A/B WMMA fragments (ISA 7.12.2):
// lane group g = lane/16 holds K in {g*8..g*8+7} U {16+g*8..16+g*8+7}
__device__ __forceinline__ int kpat(int g, int i) {
    return (i < 8) ? (g * 8 + i) : (16 + g * 8 + (i - 8));
}

// ---------------------------------------------------------------------------
// CDNA5 async copy: 16B global -> LDS, tracked by ASYNCcnt (bypasses VGPRs)
// Builtin prototype (from hipcc diagnostic): first arg is a pointer to a
// 4 x i32 vector in the global address space; LDS arg symmetric in as(3).
// ---------------------------------------------------------------------------
typedef int v4i_vs __attribute__((vector_size(16)));
typedef __attribute__((address_space(1))) v4i_vs* g_v4i_p;
typedef __attribute__((address_space(3))) v4i_vs* l_v4i_p;

__device__ __forceinline__ void async_b128(const void* g, void* l) {
#if __has_builtin(__builtin_amdgcn_global_load_async_to_lds_b128)
    __builtin_amdgcn_global_load_async_to_lds_b128(
        (g_v4i_p)(g), (l_v4i_p)(l), 0, 0);
#else
    asm volatile("global_load_async_to_lds_b128 %0, %1, off"
                 :: "v"((unsigned)(unsigned long long)(l_v4i_p)(l)),
                    "v"(g)
                 : "memory");
#endif
}

__device__ __forceinline__ void wait_async0() {
#if __has_builtin(__builtin_amdgcn_s_wait_asynccnt)
    __builtin_amdgcn_s_wait_asynccnt(0);
#else
    asm volatile("s_wait_asynccnt 0x0" ::: "memory");
#endif
}

// ---------------------------------------------------------------------------
// Generic WMMA GEMM: C[M,N] = act(A[M,K] @ B[K,N] + bias[N]), optional f16 copy
// block: 256 thr = 8 waves (4 M-rows x 2 N-cols); block tile 64x128, K-step 32
// ---------------------------------------------------------------------------
template<int RELU>
__global__ __launch_bounds__(256) void gemm_wmma_kernel(
    const float* __restrict__ A, const float* __restrict__ Bm,
    const float* __restrict__ bias, float* __restrict__ C,
    _Float16* __restrict__ Ch, int M, int N, int Kd)
{
    __shared__ _Float16 As[64][40];
    __shared__ _Float16 Bs[128][40];   // stored transposed: Bs[n][k]

    const int tid  = threadIdx.x;
    const int wave = tid >> 5, lane = tid & 31;
    const int g    = lane >> 4, ln = lane & 15;
    const int wr   = wave & 3,  wc = wave >> 2;
    const int m0   = blockIdx.y * 64, n0 = blockIdx.x * 128;

    v8f acc[4];
    #pragma unroll
    for (int t = 0; t < 4; ++t) acc[t] = {};

    for (int k0 = 0; k0 < Kd; k0 += 32) {
        __syncthreads();
        { // stage A tile 64x32 (8 contiguous floats / thread) with f32->f16
            int row = tid >> 2, col = (tid & 3) * 8;
            const float* src = A + (size_t)(m0 + row) * Kd + k0 + col;
            __builtin_prefetch(src + 32, 0, 1);   // next K tile -> global_prefetch
            #pragma unroll
            for (int i = 0; i < 8; ++i) As[row][col + i] = (_Float16)src[i];
        }
        { // stage B tile 32x128, transposed into LDS (coalesced along N)
            int kk = tid >> 3, nn = (tid & 7) * 16;
            const float* src = Bm + (size_t)(k0 + kk) * N + n0 + nn;
            #pragma unroll
            for (int i = 0; i < 16; ++i) Bs[nn + i][kk] = (_Float16)src[i];
        }
        __syncthreads();

        v16h a;
        const int arow = wr * 16 + ln;
        #pragma unroll
        for (int i = 0; i < 16; ++i) a[i] = As[arow][kpat(g, i)];

        #pragma unroll
        for (int bt = 0; bt < 4; ++bt) {
            v16h b;
            const int bn = wc * 64 + bt * 16 + ln;
            #pragma unroll
            for (int i = 0; i < 16; ++i) b[i] = Bs[bn][kpat(g, i)];
            acc[bt] = __builtin_amdgcn_wmma_f32_16x16x32_f16(
                false, a, false, b, (short)0, acc[bt], false, false);
        }
    }

    #pragma unroll
    for (int bt = 0; bt < 4; ++bt) {
        const int col = n0 + wc * 64 + bt * 16 + ln;
        const float bv = bias[col];
        #pragma unroll
        for (int r = 0; r < 8; ++r) {
            const int row = m0 + wr * 16 + g * 8 + r;
            float v = acc[bt][r] + bv;
            if (RELU) v = fmaxf(v, 0.0f);
            C[(size_t)row * N + col] = v;
            if (Ch) Ch[(size_t)row * N + col] = (_Float16)v;
        }
    }
}

// ---------------------------------------------------------------------------
// Fused attention per flattened head [64, 512, 64] (reshape-bug layout is a
// reinterpretation of contiguous [B,S,D] memory). att *= 8 (SCALE bug).
// K staged via ASYNC copies from the f16 K buffer; V transposed via VALU.
// ---------------------------------------------------------------------------
__global__ __launch_bounds__(256) void attn_kernel(
    const float* __restrict__ q, const _Float16* __restrict__ kh,
    const float* __restrict__ v, float* __restrict__ ctx)
{
    __shared__ _Float16 kl[128][72];        // [key][d]  (NT B operand), 144B rows
    __shared__ _Float16 vt[64][136];        // [d][key]  (NN B operand, transposed)
    __shared__ _Float16 pl[8][16][40];      // per-wave P tile round-trip

    const int head = blockIdx.x;
    const size_t base = (size_t)head * SEQ * DPH;
    const int tid  = threadIdx.x;
    const int wave = tid >> 5, lane = tid & 31;
    const int g    = lane >> 4, ln = lane & 15;

    for (int qt = 0; qt < 4; ++qt) {
        const int qs0 = wave * 64 + qt * 16;

        // A fragments for this 16-row q tile (K=0..31 and K=32..63)
        v16h aq0, aq1;
        {
            const float* qr = q + base + (size_t)(qs0 + ln) * DPH;
            #pragma unroll
            for (int i = 0; i < 16; ++i) {
                aq0[i] = (_Float16)qr[kpat(g, i)];
                aq1[i] = (_Float16)qr[32 + kpat(g, i)];
            }
        }

        float m[8], l[8];
        v8f o[4];
        #pragma unroll
        for (int r = 0; r < 8; ++r) { m[r] = -3.0e38f; l[r] = 0.0f; }
        #pragma unroll
        for (int t = 0; t < 4; ++t) o[t] = {};

        for (int stage = 0; stage < 4; ++stage) {
            const int ks0 = stage * 128;
            __syncthreads();
            { // K tile: 128 rows x 128B, pure async byte copy (already f16)
                int row = tid >> 1, half = tid & 1;
                const _Float16* src = kh + base + (size_t)(ks0 + row) * DPH + half * 32;
                #pragma unroll
                for (int j = 0; j < 4; ++j)
                    async_b128(src + j * 8, &kl[row][half * 32 + j * 8]);
            }
            { // V tile: f32 -> f16 with transpose (VALU staging)
                int row = tid >> 1, c0 = (tid & 1) * 32;
                const float* vr = v + base + (size_t)(ks0 + row) * DPH + c0;
                #pragma unroll
                for (int i = 0; i < 32; ++i)
                    vt[c0 + i][row] = (_Float16)vr[i];
            }
            wait_async0();
            __syncthreads();

            for (int jb = 0; jb < 4; ++jb) {
                const int jl0 = jb * 32;
                v8f s[2];
                s[0] = {}; s[1] = {};
                #pragma unroll
                for (int h = 0; h < 2; ++h) {
                    v16h b0, b1;
                    const int bn = jl0 + h * 16 + ln;
                    #pragma unroll
                    for (int i = 0; i < 16; ++i) {
                        b0[i] = kl[bn][kpat(g, i)];
                        b1[i] = kl[bn][32 + kpat(g, i)];
                    }
                    s[h] = __builtin_amdgcn_wmma_f32_16x16x32_f16(
                        false, aq0, false, b0, (short)0, s[h], false, false);
                    s[h] = __builtin_amdgcn_wmma_f32_16x16x32_f16(
                        false, aq1, false, b1, (short)0, s[h], false, false);
                }
                s[0] = s[0] * 8.0f;  // replicates torch SCALE bug (multiply)
                s[1] = s[1] * 8.0f;

                float cm[8], p0[8], p1[8], rs[8];
                #pragma unroll
                for (int r = 0; r < 8; ++r) cm[r] = fmaxf(s[0][r], s[1][r]);
                #pragma unroll
                for (int mask = 1; mask <= 8; mask <<= 1)
                    #pragma unroll
                    for (int r = 0; r < 8; ++r)
                        cm[r] = fmaxf(cm[r], __shfl_xor(cm[r], mask, 32));
                #pragma unroll
                for (int r = 0; r < 8; ++r) {
                    float nm   = fmaxf(m[r], cm[r]);
                    float corr = __expf(m[r] - nm);
                    p0[r] = __expf(s[0][r] - nm);
                    p1[r] = __expf(s[1][r] - nm);
                    rs[r] = p0[r] + p1[r];
                    m[r]  = nm;
                    l[r]  = l[r] * corr;
                    #pragma unroll
                    for (int t = 0; t < 4; ++t) o[t][r] = o[t][r] * corr;
                }
                #pragma unroll
                for (int mask = 1; mask <= 8; mask <<= 1)
                    #pragma unroll
                    for (int r = 0; r < 8; ++r)
                        rs[r] += __shfl_xor(rs[r], mask, 32);
                #pragma unroll
                for (int r = 0; r < 8; ++r) l[r] += rs[r];

                // C-layout -> A-layout via per-wave LDS
                #pragma unroll
                for (int r = 0; r < 8; ++r) {
                    pl[wave][r + 8 * g][ln]      = (_Float16)p0[r];
                    pl[wave][r + 8 * g][16 + ln] = (_Float16)p1[r];
                }
                asm volatile("s_wait_dscnt 0" ::: "memory");   // wave-local LDS RAW

                v16h ap;
                #pragma unroll
                for (int i = 0; i < 16; ++i) ap[i] = pl[wave][ln][kpat(g, i)];

                #pragma unroll
                for (int t = 0; t < 4; ++t) {
                    v16h bvf;
                    const int bn = t * 16 + ln;
                    #pragma unroll
                    for (int i = 0; i < 16; ++i) bvf[i] = vt[bn][jl0 + kpat(g, i)];
                    o[t] = __builtin_amdgcn_wmma_f32_16x16x32_f16(
                        false, ap, false, bvf, (short)0, o[t], false, false);
                }
            }
        }

        #pragma unroll
        for (int t = 0; t < 4; ++t)
            #pragma unroll
            for (int r = 0; r < 8; ++r) {
                const int row = qs0 + g * 8 + r;
                ctx[base + (size_t)row * DPH + t * 16 + ln] = o[t][r] / l[r];
            }
    }
}

// ---------------------------------------------------------------------------
// LayerNorm(resid + delta) * g + b   (one block per token row)
// ---------------------------------------------------------------------------
__global__ __launch_bounds__(256) void ln_kernel(
    const float* __restrict__ resid, const float* __restrict__ delta,
    const float* __restrict__ gam, const float* __restrict__ bet,
    float* __restrict__ out)
{
    __shared__ float red[16];
    const int row = blockIdx.x, tid = threadIdx.x;
    const size_t base = (size_t)row * D_MODEL;

    float y0 = resid[base + tid]       + delta[base + tid];
    float y1 = resid[base + 256 + tid] + delta[base + 256 + tid];
    float s  = y0 + y1;
    float sq = y0 * y0 + y1 * y1;
    #pragma unroll
    for (int mask = 16; mask >= 1; mask >>= 1) {
        s  += __shfl_xor(s,  mask, 32);
        sq += __shfl_xor(sq, mask, 32);
    }
    const int wave = tid >> 5, lane = tid & 31;
    if (lane == 0) { red[wave] = s; red[8 + wave] = sq; }
    __syncthreads();
    float ts = 0.0f, tq = 0.0f;
    #pragma unroll
    for (int w = 0; w < 8; ++w) { ts += red[w]; tq += red[8 + w]; }
    const float mu  = ts * (1.0f / D_MODEL);
    const float var = tq * (1.0f / D_MODEL) - mu * mu;
    const float rs  = rsqrtf(var + 1e-5f);
    out[base + tid]       = (y0 - mu) * rs * gam[tid]       + bet[tid];
    out[base + 256 + tid] = (y1 - mu) * rs * gam[256 + tid] + bet[256 + tid];
}

// ---------------------------------------------------------------------------
// halfc[k] = 0.5 * ||codebook[k]||^2, plus f16 codebook copy for async staging
// ---------------------------------------------------------------------------
__global__ __launch_bounds__(128) void cnorm_kernel(
    const float* __restrict__ cb, float* __restrict__ halfc,
    _Float16* __restrict__ cbh)
{
    __shared__ float red[4];
    const int kk = blockIdx.x, tid = threadIdx.x;
    const float* row = cb + (size_t)kk * D_MODEL + tid * 4;
    _Float16* hrow = cbh + (size_t)kk * D_MODEL + tid * 4;
    float s = 0.0f;
    #pragma unroll
    for (int i = 0; i < 4; ++i) {
        const float x = row[i];
        hrow[i] = (_Float16)x;
        s += x * x;
    }
    #pragma unroll
    for (int mask = 16; mask >= 1; mask >>= 1) s += __shfl_xor(s, mask, 32);
    if ((tid & 31) == 0) red[tid >> 5] = s;
    __syncthreads();
    if (tid == 0) halfc[kk] = 0.5f * (red[0] + red[1] + red[2] + red[3]);
}

// f32 -> f16 bulk convert (for encoder output feeding the VQ GEMM)
__global__ __launch_bounds__(256) void h16_convert_kernel(
    const float* __restrict__ src, _Float16* __restrict__ dst)
{
    const size_t i0 = ((size_t)blockIdx.x * 256 + threadIdx.x) * 4;
    #pragma unroll
    for (int i = 0; i < 4; ++i) dst[i0 + i] = (_Float16)src[i0 + i];
}

// ---------------------------------------------------------------------------
// Fused VQ: argmax_k ( z·c_k - 0.5||c_k||^2 )  == argmin distance.
// Both operands pre-converted f16 -> tiles staged with ASYNC LDS copies.
// ---------------------------------------------------------------------------
__global__ __launch_bounds__(256) void vq_kernel(
    const _Float16* __restrict__ zh, const _Float16* __restrict__ cbh,
    const float* __restrict__ halfc, int* __restrict__ idx_out)
{
    __shared__ _Float16 As[64][40];    // 80B rows -> 16B-aligned b128 chunks
    __shared__ _Float16 Bs[128][40];
    __shared__ float rv[64];
    __shared__ int   ri[64];

    const int tid  = threadIdx.x;
    const int wave = tid >> 5, lane = tid & 31;
    const int g    = lane >> 4, ln = lane & 15;
    const int wr   = wave & 3,  wc = wave >> 2;
    const int m0   = blockIdx.x * 64;

    float bestv[8];
    int   besti[8];
    #pragma unroll
    for (int r = 0; r < 8; ++r) { bestv[r] = -3.0e38f; besti[r] = 0; }

    for (int cb0 = 0; cb0 < KCODES; cb0 += 128) {
        v8f acc[4];
        #pragma unroll
        for (int t = 0; t < 4; ++t) acc[t] = {};

        for (int k0 = 0; k0 < D_MODEL; k0 += 32) {
            __syncthreads();
            { // z tile: 64 rows x 64B = 256 b128 chunks, 1/thread (async)
                int row = tid >> 2, c = tid & 3;
                async_b128(zh + (size_t)(m0 + row) * D_MODEL + k0 + c * 8,
                           &As[row][c * 8]);
            }
            { // codebook tile: 128 rows x 64B = 512 chunks, 2/thread (async)
                #pragma unroll
                for (int j = 0; j < 2; ++j) {
                    int chunk = tid * 2 + j;
                    int n = chunk >> 2, c = chunk & 3;
                    async_b128(cbh + (size_t)(cb0 + n) * D_MODEL + k0 + c * 8,
                               &Bs[n][c * 8]);
                }
            }
            wait_async0();
            __syncthreads();

            v16h a;
            const int arow = wr * 16 + ln;
            #pragma unroll
            for (int i = 0; i < 16; ++i) a[i] = As[arow][kpat(g, i)];
            #pragma unroll
            for (int bt = 0; bt < 4; ++bt) {
                v16h b;
                const int bn = wc * 64 + bt * 16 + ln;
                #pragma unroll
                for (int i = 0; i < 16; ++i) b[i] = Bs[bn][kpat(g, i)];
                acc[bt] = __builtin_amdgcn_wmma_f32_16x16x32_f16(
                    false, a, false, b, (short)0, acc[bt], false, false);
            }
        }

        #pragma unroll
        for (int bt = 0; bt < 4; ++bt) {
            const int code = cb0 + wc * 64 + bt * 16 + ln;
            const float hc = halfc[code];
            #pragma unroll
            for (int r = 0; r < 8; ++r) {
                const float sc = acc[bt][r] - hc;
                if (sc > bestv[r]) { bestv[r] = sc; besti[r] = code; }
            }
        }
    }

    // cross-lane argmax (ties -> smaller code index, matches argmin-first)
    #pragma unroll
    for (int mask = 1; mask <= 8; mask <<= 1)
        #pragma unroll
        for (int r = 0; r < 8; ++r) {
            float ov = __shfl_xor(bestv[r], mask, 32);
            int   oi = __shfl_xor(besti[r], mask, 32);
            if (ov > bestv[r] || (ov == bestv[r] && oi < besti[r])) {
                bestv[r] = ov; besti[r] = oi;
            }
        }
    __syncthreads();
    if (wc == 0 && ln == 0)
        #pragma unroll
        for (int r = 0; r < 8; ++r) {
            const int row = wr * 16 + g * 8 + r;
            rv[row] = bestv[r]; ri[row] = besti[r];
        }
    __syncthreads();
    if (wc == 1 && ln == 0)
        #pragma unroll
        for (int r = 0; r < 8; ++r) {
            const int row = wr * 16 + g * 8 + r;
            float v2 = rv[row]; int i2 = ri[row];
            int fin = (v2 > bestv[r] || (v2 == bestv[r] && i2 < besti[r]))
                          ? i2 : besti[r];
            idx_out[m0 + row] = fin;
        }
}

// ---------------------------------------------------------------------------
// Gather codebook rows, write zq = z + (q - z) (faithful straight-through),
// accumulate SSE + histogram, emit idx as float.
// ---------------------------------------------------------------------------
__global__ __launch_bounds__(256) void gather_kernel(
    const float* __restrict__ z, const float* __restrict__ cb,
    const int* __restrict__ idx, float* __restrict__ zq,
    float* __restrict__ idx_f, float* __restrict__ sse_acc,
    int* __restrict__ hist)
{
    __shared__ float red[8];
    const int row = blockIdx.x, tid = threadIdx.x;
    const int id = idx[row];
    const size_t base = (size_t)row * D_MODEL;
    const size_t cbase = (size_t)id * D_MODEL;

    float s = 0.0f;
    #pragma unroll
    for (int h = 0; h < 2; ++h) {
        const int j = tid + h * 256;
        const float qv = cb[cbase + j];
        const float zv = z[base + j];
        zq[base + j] = zv + (qv - zv);
        const float d = qv - zv;
        s += d * d;
    }
    #pragma unroll
    for (int mask = 16; mask >= 1; mask >>= 1) s += __shfl_xor(s, mask, 32);
    if ((tid & 31) == 0) red[tid >> 5] = s;
    __syncthreads();
    if (tid == 0) {
        float t = 0.0f;
        #pragma unroll
        for (int w = 0; w < 8; ++w) t += red[w];
        atomicAdd(sse_acc, t);
        atomicAdd(&hist[id], 1);
        idx_f[row] = (float)id;
    }
}

__global__ __launch_bounds__(256) void finalize_kernel(
    const int* __restrict__ hist, const float* __restrict__ sse,
    float* __restrict__ out_loss, float* __restrict__ out_perp)
{
    __shared__ float red[8];
    const int tid = threadIdx.x;
    float s = 0.0f;
    for (int kk = tid; kk < KCODES; kk += 256) {
        const float p = (float)hist[kk] * (1.0f / NTOK);
        s += p * logf(p + 1e-10f);
    }
    #pragma unroll
    for (int mask = 16; mask >= 1; mask >>= 1) s += __shfl_xor(s, mask, 32);
    if ((tid & 31) == 0) red[tid >> 5] = s;
    __syncthreads();
    if (tid == 0) {
        float t = 0.0f;
        #pragma unroll
        for (int w = 0; w < 8; ++w) t += red[w];
        *out_loss = 0.25f * (*sse) / ((float)NTOK * (float)D_MODEL);
        *out_perp = expf(-t);
    }
}

// ---------------------------------------------------------------------------
extern "C" void kernel_launch(void* const* d_in, const int* in_sizes, int n_in,
                              void* d_out, int out_size, void* d_ws, size_t ws_size,
                              hipStream_t stream) {
    const float* X    = (const float*)d_in[0];
    const float* Wq   = (const float*)d_in[1];  const float* bq = (const float*)d_in[2];
    const float* Wk   = (const float*)d_in[3];  const float* bk = (const float*)d_in[4];
    const float* Wv   = (const float*)d_in[5];  const float* bv = (const float*)d_in[6];
    const float* Wo   = (const float*)d_in[7];  const float* bo = (const float*)d_in[8];
    const float* ln1g = (const float*)d_in[9];  const float* ln1b = (const float*)d_in[10];
    const float* W1   = (const float*)d_in[11]; const float* b1 = (const float*)d_in[12];
    const float* W2   = (const float*)d_in[13]; const float* b2 = (const float*)d_in[14];
    const float* ln2g = (const float*)d_in[15]; const float* ln2b = (const float*)d_in[16];
    const float* cb   = (const float*)d_in[17];

    const size_t P = (size_t)NTOK * D_MODEL;   // 2,097,152 floats
    float* W    = (float*)d_ws;
    float* xbuf = W;             // P
    float* xb   = W + 1 * P;     // P
    float* qb   = W + 2 * P;
    float* kb   = W + 3 * P;
    float* vb   = W + 4 * P;
    float* ctx  = W + 5 * P;
    float* tmp  = W + 6 * P;
    float* hid  = W + 7 * P;     // 4*P
    float* halfc = W + 11 * P;   // 16384
    float* sse   = halfc + KCODES;
    int*   idxb  = (int*)(sse + 1);
    int*   hist  = idxb + NTOK;
    _Float16* kh  = (_Float16*)(W + 12 * P);        // P halfs
    _Float16* zh  = (_Float16*)(W + 12 * P) + P;    // P halfs
    _Float16* cbh = (_Float16*)(W + 13 * P);        // KCODES*D halfs (2P floats)

    float* out_zq   = (float*)d_out;
    float* out_idx  = out_zq + P;
    float* out_loss = out_idx + NTOK;
    float* out_perp = out_loss + 1;

    const dim3 blk(256);
    const dim3 gD(D_MODEL / 128, NTOK / 64);    // N=512 GEMMs
    const dim3 gF(FFN_DIM / 128, NTOK / 64);    // N=2048 GEMM
    _Float16* nullh = nullptr;

    const float* xin = X;
    for (int l = 0; l < N_LAYERS; ++l) {
        const size_t dd = (size_t)D_MODEL * D_MODEL;
        const size_t df = (size_t)D_MODEL * FFN_DIM;
        gemm_wmma_kernel<0><<<gD, blk, 0, stream>>>(xin, Wq + l * dd, bq + l * D_MODEL, qb, nullh, NTOK, D_MODEL, D_MODEL);
        gemm_wmma_kernel<0><<<gD, blk, 0, stream>>>(xin, Wk + l * dd, bk + l * D_MODEL, kb, kh,    NTOK, D_MODEL, D_MODEL);
        gemm_wmma_kernel<0><<<gD, blk, 0, stream>>>(xin, Wv + l * dd, bv + l * D_MODEL, vb, nullh, NTOK, D_MODEL, D_MODEL);
        attn_kernel<<<NHEAD_T, blk, 0, stream>>>(qb, kh, vb, ctx);
        gemm_wmma_kernel<0><<<gD, blk, 0, stream>>>(ctx, Wo + l * dd, bo + l * D_MODEL, tmp, nullh, NTOK, D_MODEL, D_MODEL);
        ln_kernel<<<NTOK, blk, 0, stream>>>(xin, tmp, ln1g + l * D_MODEL, ln1b + l * D_MODEL, xb);
        gemm_wmma_kernel<1><<<gF, blk, 0, stream>>>(xb, W1 + l * df, b1 + l * FFN_DIM, hid, nullh, NTOK, FFN_DIM, D_MODEL);
        gemm_wmma_kernel<0><<<gD, blk, 0, stream>>>(hid, W2 + l * df, b2 + l * D_MODEL, tmp, nullh, NTOK, D_MODEL, FFN_DIM);
        ln_kernel<<<NTOK, blk, 0, stream>>>(xb, tmp, ln2g + l * D_MODEL, ln2b + l * D_MODEL, xbuf);
        xin = xbuf;
    }

    (void)hipMemsetAsync(sse, 0, sizeof(float), stream);
    (void)hipMemsetAsync(hist, 0, KCODES * sizeof(int), stream);
    cnorm_kernel<<<KCODES, dim3(128), 0, stream>>>(cb, halfc, cbh);
    h16_convert_kernel<<<P / 1024, blk, 0, stream>>>(xin, zh);
    vq_kernel<<<NTOK / 64, blk, 0, stream>>>(zh, cbh, halfc, idxb);
    gather_kernel<<<NTOK, blk, 0, stream>>>(xin, cb, idxb, out_zq, out_idx, sse, hist);
    finalize_kernel<<<1, blk, 0, stream>>>(hist, sse, out_loss, out_perp);
}
